// GNN_EBM_Layer_01_71347996721296
// MI455X (gfx1250) — matrix operand order repaired and checked by reading, not compiled, not run on hardware
//
#include <hip/hip_runtime.h>

// ---------------------------------------------------------------------------
// GNN EBM layer for MI455X (gfx1250, wave32, WMMA).
// Fused formulation:
//   e  = x_2nd @ W_edge^T + (edge_i+edge_j) @ (W_edge·W_mlp)^T + (W_edge·b_mlp + b_edge)
//   x  = (x_1st + 0.5*scatter(pairsum(x_2nd))) @ W_node^T + b_node
// All GEMMs run as fp32-emulated bf16 WMMA (hi/lo split, 3 products), f32 accum.
// ---------------------------------------------------------------------------

typedef __attribute__((ext_vector_type(16))) __bf16 v16bf;
typedef __attribute__((ext_vector_type(8)))  float  v8f;

#define B_  8
#define T_  1024
#define M_  16384
#define D_  128

// ---- workspace layout (bytes) ----
static constexpr size_t UPD_FLOATS = (size_t)B_ * T_ * 2 * D_;     // 2,097,152
static constexpr size_t OFF_UPD = 0;                                // 8 MB f32 accum
static constexpr size_t OFF_WN  = UPD_FLOATS * 4;                   // W_node frags: 32KB hi + 32KB lo
static constexpr size_t OFF_WE  = OFF_WN + 65536;                   // W_edge frags
static constexpr size_t OFF_WC  = OFF_WE + 65536;                   // Wc = W_edge·W_mlp frags
static constexpr size_t OFF_BN  = OFF_WC + 65536;                   // b_node (128 f32)
static constexpr size_t OFF_BC  = OFF_BN + 512;                     // combined edge bias (128 f32)

__device__ __forceinline__ v8f wmma_bf16(v16bf a, v16bf b, v8f c) {
  // emits v_wmma_f32_16x16x32_bf16
  return __builtin_amdgcn_wmma_f32_16x16x32_bf16(false, a, false, b, (short)0, c,
                                                 false, false);
}

// split a float4 into bf16 hi + residual lo at element positions [base..base+3]
__device__ __forceinline__ void split4(float4 f, v16bf& hi, v16bf& lo, int base) {
  float v[4] = {f.x, f.y, f.z, f.w};
#pragma unroll
  for (int i = 0; i < 4; ++i) {
    __bf16 h = (__bf16)v[i];
    hi[base + i] = h;
    lo[base + i] = (__bf16)(v[i] - (float)h);
  }
}

__device__ __forceinline__ float4 add4(float4 a, float4 b) {
  return make_float4(a.x + b.x, a.y + b.y, a.z + b.z, a.w + b.w);
}

// ---------------------------------------------------------------------------
__global__ __launch_bounds__(256) void zero_upd(float4* __restrict__ p) {
  p[(size_t)blockIdx.x * 256 + threadIdx.x] = make_float4(0.f, 0.f, 0.f, 0.f);
}

// ---------------------------------------------------------------------------
// Weight prep: Wc = W_edge·W_mlp, combined bias, and pack all weights into the
// WMMA B-matrix fragment order:  frag[((oc*4+kc)*32+lane)*16+i] =
//   W[ oc*16 + (lane&15) ][ kc*32 + (lane>>4)*8 + (i&7) + (i>>3)*16 ]   (lane = column of B)
// stored as bf16 hi (16384) then bf16 lo (16384) per matrix.
// ---------------------------------------------------------------------------
__global__ __launch_bounds__(256) void prep_kernel(
    const float* __restrict__ W_node, const float* __restrict__ b_node,
    const float* __restrict__ W_edge, const float* __restrict__ b_edge,
    const float* __restrict__ W_mlp,  const float* __restrict__ b_mlp,
    char* __restrict__ ws) {
  __shared__ float sWc[128 * 128];  // 64 KB
  const int tid = threadIdx.x;

  for (int idx = tid; idx < 128 * 128; idx += 256) {
    int o = idx >> 7, d = idx & 127;
    float s = 0.f;
    for (int h = 0; h < 128; ++h) s += W_edge[o * 128 + h] * W_mlp[h * 128 + d];
    sWc[idx] = s;
  }
  float* bN = (float*)(ws + OFF_BN);
  float* bC = (float*)(ws + OFF_BC);
  for (int o = tid; o < 128; o += 256) {
    float s = b_edge[o];
    for (int h = 0; h < 128; ++h) s += W_edge[o * 128 + h] * b_mlp[h];
    bC[o] = s;
    bN[o] = b_node[o];
  }
  __syncthreads();

  __bf16* WN = (__bf16*)(ws + OFF_WN);
  __bf16* WE = (__bf16*)(ws + OFF_WE);
  __bf16* WC = (__bf16*)(ws + OFF_WC);
  for (int fi = tid; fi < 16384; fi += 256) {
    int i = fi & 15, lane = (fi >> 4) & 31, kc = (fi >> 9) & 3, oc = fi >> 11;
    int o = oc * 16 + (lane & 15);
    int d = kc * 32 + (lane >> 4) * 8 + (i & 7) + (i >> 3) * 16;
    float vn = W_node[o * 128 + d];
    float ve = W_edge[o * 128 + d];
    float vc = sWc[o * 128 + d];
    __bf16 h;
    h = (__bf16)vn; WN[fi] = h; WN[16384 + fi] = (__bf16)(vn - (float)h);
    h = (__bf16)ve; WE[fi] = h; WE[16384 + fi] = (__bf16)(ve - (float)h);
    h = (__bf16)vc; WC[fi] = h; WC[16384 + fi] = (__bf16)(vc - (float)h);
  }
}

// ---------------------------------------------------------------------------
// Edge kernel: 4096 blocks (8 batches x 512), 256 threads = 8 waves.
// Block covers 128 rows of the (B*M*4)-row edge GEMM = 32 edges.
//  - phase 1: pair-sum scatter of x_2nd into upd (atomic f32, x0.5)
//  - phase 2: per wave a 16x128 output stripe: 6 WMMAs per (kc,oc).
// 128 KB dynamic LDS holds W_edge + Wc fragments (hi/lo).
// ---------------------------------------------------------------------------
__global__ __launch_bounds__(256) void edge_kernel(
    const float* __restrict__ x1, const float* __restrict__ x2,
    const int* __restrict__ edge, char* __restrict__ ws,
    float* __restrict__ e_out) {
  extern __shared__ char smem[];  // [WEhi 32K | WElo 32K | WChi 32K | WClo 32K]
  {
    const uint4* src = (const uint4*)(ws + OFF_WE);  // WE then WC, contiguous
    uint4* dst = (uint4*)smem;
    for (int i = threadIdx.x; i < 8192; i += 256) dst[i] = src[i];
  }
  __syncthreads();

  const int bid  = blockIdx.x;
  const int b    = bid >> 9;
  const int rblk = (bid & 511) << 7;  // row base within this batch [0,65536)
  const int tid  = threadIdx.x;
  const int lane = tid & 31, wave = tid >> 5;
  float* upd = (float*)(ws + OFF_UPD) + (size_t)b * T_ * 2 * D_;

  // ---------- scatter phase: 32 edges, 8 threads per edge ----------
  {
    const int el = tid >> 3;          // local edge 0..31
    const int dc = (tid & 7) << 4;    // 16-float d chunk
    const int em = (rblk >> 2) + el;
    const int ni = edge[em], no = edge[M_ + em];
    const float* p = x2 + ((size_t)b * M_ + em) * 4 * D_;
#pragma unroll
    for (int s = 0; s < 2; ++s) {
      float* uo = upd + ((size_t)no * 2 + s) * D_ + dc;
      float* ui = upd + ((size_t)ni * 2 + s) * D_ + dc;
#pragma unroll
      for (int d = 0; d < 16; ++d) {
        float v = 0.5f * (p[(2 * s) * D_ + dc + d] + p[(2 * s + 1) * D_ + dc + d]);
        atomicAdd(uo + d, v);
        atomicAdd(ui + d, v);
      }
    }
  }

  // ---------- GEMM phase ----------
  const int hv  = lane >> 4;     // half-wave selects K sub-block
  const int col = lane & 15;     // A row / B-C-D column
  const int row = rblk + wave * 16 + col;  // row within batch (m*4+k)
  const int em  = row >> 2, k = row & 3;
  const int ni  = edge[em], no = edge[M_ + em];
  const float* pX2 = x2 + (((size_t)b * M_ + em) * 4 + k) * D_;
  const float* pGi = x1 + (((size_t)b * T_ + ni) * 2 + (k >> 1)) * D_;
  const float* pGj = x1 + (((size_t)b * T_ + no) * 2 + (k & 1)) * D_;

  v8f acc[8];
#pragma unroll
  for (int oc = 0; oc < 8; ++oc)
#pragma unroll
    for (int g = 0; g < 8; ++g) acc[oc][g] = 0.f;

  const v16bf* fEh = (const v16bf*)(smem);
  const v16bf* fEl = (const v16bf*)(smem + 32768);
  const v16bf* fCh = (const v16bf*)(smem + 65536);
  const v16bf* fCl = (const v16bf*)(smem + 98304);

#pragma unroll
  for (int kc = 0; kc < 4; ++kc) {
    const int d0 = kc * 32 + hv * 8;  // A layout: K = {d0..d0+7, d0+16..d0+23}
    v16bf xh, xl, gh, gl;
    split4(*(const float4*)(pX2 + d0),      xh, xl, 0);
    split4(*(const float4*)(pX2 + d0 + 4),  xh, xl, 4);
    split4(*(const float4*)(pX2 + d0 + 16), xh, xl, 8);
    split4(*(const float4*)(pX2 + d0 + 20), xh, xl, 12);
    split4(add4(*(const float4*)(pGi + d0),      *(const float4*)(pGj + d0)),      gh, gl, 0);
    split4(add4(*(const float4*)(pGi + d0 + 4),  *(const float4*)(pGj + d0 + 4)),  gh, gl, 4);
    split4(add4(*(const float4*)(pGi + d0 + 16), *(const float4*)(pGj + d0 + 16)), gh, gl, 8);
    split4(add4(*(const float4*)(pGi + d0 + 20), *(const float4*)(pGj + d0 + 20)), gh, gl, 12);

#pragma unroll
    for (int oc = 0; oc < 8; ++oc) {
      const int fidx = (oc * 4 + kc) * 32 + lane;
      v16bf bh = fEh[fidx], bl = fEl[fidx];
      acc[oc] = wmma_bf16(xh, bh, acc[oc]);   // hi*hi
      acc[oc] = wmma_bf16(xh, bl, acc[oc]);   // hi*lo
      acc[oc] = wmma_bf16(xl, bh, acc[oc]);   // lo*hi
      v16bf ch = fCh[fidx], cl = fCl[fidx];
      acc[oc] = wmma_bf16(gh, ch, acc[oc]);
      acc[oc] = wmma_bf16(gh, cl, acc[oc]);
      acc[oc] = wmma_bf16(gl, ch, acc[oc]);
    }
  }

  // ---------- writeout:  C/D layout: VGPR g -> row g+8*hv, col = lane&15 ----
  const float* bC = (const float*)(ws + OFF_BC);
  const size_t gr0 = (size_t)b * ((size_t)M_ * 4) + rblk + wave * 16;
#pragma unroll
  for (int oc = 0; oc < 8; ++oc) {
    const float bias = bC[oc * 16 + col];
#pragma unroll
    for (int g = 0; g < 8; ++g) {
      const size_t r = gr0 + g + 8 * hv;
      e_out[r * D_ + oc * 16 + col] = acc[oc][g] + bias;
    }
  }
}

// ---------------------------------------------------------------------------
// Node kernel: rows = B*T*2 = 16384.  x = (x_1st + upd) @ W_node^T + b_node.
// 128 blocks x 256 threads; 64 KB dynamic LDS for W_node fragments.
// ---------------------------------------------------------------------------
__global__ __launch_bounds__(256) void node_kernel(
    const float* __restrict__ x1, char* __restrict__ ws,
    float* __restrict__ x_out) {
  extern __shared__ char smem[];  // [WNhi 32K | WNlo 32K]
  {
    const uint4* src = (const uint4*)(ws + OFF_WN);
    uint4* dst = (uint4*)smem;
    for (int i = threadIdx.x; i < 4096; i += 256) dst[i] = src[i];
  }
  __syncthreads();

  const int lane = threadIdx.x & 31, wave = threadIdx.x >> 5;
  const int hv = lane >> 4, col = lane & 15;
  const int row = blockIdx.x * 128 + wave * 16 + col;  // [0, 16384)
  const float* pA = x1 + (size_t)row * D_;
  const float* pU = (const float*)(ws + OFF_UPD) + (size_t)row * D_;

  v8f acc[8];
#pragma unroll
  for (int oc = 0; oc < 8; ++oc)
#pragma unroll
    for (int g = 0; g < 8; ++g) acc[oc][g] = 0.f;

  const v16bf* fh = (const v16bf*)(smem);
  const v16bf* fl = (const v16bf*)(smem + 32768);

#pragma unroll
  for (int kc = 0; kc < 4; ++kc) {
    const int d0 = kc * 32 + hv * 8;
    v16bf ah, al;
    split4(add4(*(const float4*)(pA + d0),      *(const float4*)(pU + d0)),      ah, al, 0);
    split4(add4(*(const float4*)(pA + d0 + 4),  *(const float4*)(pU + d0 + 4)),  ah, al, 4);
    split4(add4(*(const float4*)(pA + d0 + 16), *(const float4*)(pU + d0 + 16)), ah, al, 8);
    split4(add4(*(const float4*)(pA + d0 + 20), *(const float4*)(pU + d0 + 20)), ah, al, 12);
#pragma unroll
    for (int oc = 0; oc < 8; ++oc) {
      const int fidx = (oc * 4 + kc) * 32 + lane;
      v16bf bh = fh[fidx], bl = fl[fidx];
      acc[oc] = wmma_bf16(ah, bh, acc[oc]);
      acc[oc] = wmma_bf16(ah, bl, acc[oc]);
      acc[oc] = wmma_bf16(al, bh, acc[oc]);
    }
  }

  const float* bN = (const float*)(ws + OFF_BN);
  const size_t r0 = (size_t)blockIdx.x * 128 + wave * 16;
#pragma unroll
  for (int oc = 0; oc < 8; ++oc) {
    const float bias = bN[oc * 16 + col];
#pragma unroll
    for (int g = 0; g < 8; ++g) {
      const size_t r = r0 + g + 8 * hv;
      x_out[r * D_ + oc * 16 + col] = acc[oc][g] + bias;
    }
  }
}

// ---------------------------------------------------------------------------
extern "C" void kernel_launch(void* const* d_in, const int* in_sizes, int n_in,
                              void* d_out, int out_size, void* d_ws, size_t ws_size,
                              hipStream_t stream) {
  (void)in_sizes; (void)n_in; (void)out_size; (void)ws_size;
  const float* x1     = (const float*)d_in[0];
  const float* x2     = (const float*)d_in[1];
  const int*   edge   = (const int*)d_in[2];
  const float* W_node = (const float*)d_in[3];
  const float* b_node = (const float*)d_in[4];
  const float* W_edge = (const float*)d_in[5];
  const float* b_edge = (const float*)d_in[6];
  const float* W_mlp  = (const float*)d_in[7];
  const float* b_mlp  = (const float*)d_in[8];

  float* out   = (float*)d_out;
  float* x_out = out;                                  // (B,T,2,128)
  float* e_out = out + (size_t)B_ * T_ * 2 * 128;      // (B,M,4,128)
  char*  ws    = (char*)d_ws;

  // 1) clear scatter accumulator (8 MB)
  zero_upd<<<dim3(UPD_FLOATS / 4 / 256), dim3(256), 0, stream>>>(
      (float4*)(ws + OFF_UPD));
  // 2) fold weights, pack bf16 hi/lo WMMA fragments
  prep_kernel<<<dim3(1), dim3(256), 0, stream>>>(W_node, b_node, W_edge, b_edge,
                                                 W_mlp, b_mlp, ws);
  // 3) fused edge pass: scatter + e GEMM (524288 x 128 x 128, twice, fused)
  edge_kernel<<<dim3(B_ * 512), dim3(256), 131072, stream>>>(x1, x2, edge, ws,
                                                             e_out);
  // 4) node GEMM on (x_1st + upd)
  node_kernel<<<dim3(128), dim3(256), 65536, stream>>>(x1, ws, x_out);
}
